// DistanceAwareMultiHeadAttention_13718125543677
// MI455X (gfx1250) — matrix operand order repaired and checked by reading, not compiled, not run on hardware
//
#include <hip/hip_runtime.h>
#include <hip/hip_bf16.h>

typedef __bf16 bf16_t;
typedef __attribute__((ext_vector_type(16))) __bf16 v16bf;
typedef __attribute__((ext_vector_type(8)))  __bf16 v8bf;
typedef __attribute__((ext_vector_type(8)))  float  v8f;

#define HID  512
#define NSEQ 512
#define BATCH 8
#define NH   8
#define HD   64
#define DD   64

// ---------- helpers ----------
static __device__ __forceinline__ v8bf load8bf(const bf16_t* p) {
    return *reinterpret_cast<const v8bf*>(p);
}
static __device__ __forceinline__ v16bf cat16(v8bf lo, v8bf hi) {
    return __builtin_shufflevector(lo, hi, 0,1,2,3,4,5,6,7,8,9,10,11,12,13,14,15);
}
static __device__ __forceinline__ v8f wmma_bf16(v16bf a, v16bf b, v8f c) {
    // D = A(16x32 bf16) x B(32x16 bf16) + C(16x16 f32)
    return __builtin_amdgcn_wmma_f32_16x16x32_bf16(false, a, false, b, (short)0, c, false, false);
}
// A-fragment (16-bit, 16x32): lane m = lane%16, kh = lane/16,
// element e -> K = (e>>3)*16 + kh*8 + (e&7)  => two contiguous groups of 8.
static __device__ __forceinline__ v16bf a_frag_bf16(const bf16_t* rowp, int kh) {
    return cat16(load8bf(rowp + kh*8), load8bf(rowp + 16 + kh*8));
}
// B-fragment (16-bit, 32x16): lane col = lane%16, element e -> K = (e>>3)*16 + kh*8 + (e&7) (same as A).
static __device__ __forceinline__ v16bf b_frag_bf16(const bf16_t* colp, int kh) {
    return cat16(load8bf(colp + kh*8), load8bf(colp + 16 + kh*8));
}
static __device__ __forceinline__ v16bf a_frag_f32(const float* rowp, int kh) {
    float4 x0 = *reinterpret_cast<const float4*>(rowp + kh*8);
    float4 x1 = *reinterpret_cast<const float4*>(rowp + kh*8 + 4);
    float4 y0 = *reinterpret_cast<const float4*>(rowp + 16 + kh*8);
    float4 y1 = *reinterpret_cast<const float4*>(rowp + 16 + kh*8 + 4);
    float t[16] = {x0.x,x0.y,x0.z,x0.w, x1.x,x1.y,x1.z,x1.w,
                   y0.x,y0.y,y0.z,y0.w, y1.x,y1.y,y1.z,y1.w};
    v16bf a;
    #pragma unroll
    for (int e = 0; e < 16; ++e) a[e] = (__bf16)t[e];
    return a;
}

// ---------- K0: transpose + bf16-convert the 4 weight matrices ----------
// wT[w][c][k] = bf16(W_w[k][c]);  w: 0=Wq 1=Wk 2=Wv 3=Wo
__global__ void __launch_bounds__(256) k_transpose_w(
    const float* __restrict__ Wq, const float* __restrict__ Wk,
    const float* __restrict__ Wv, const float* __restrict__ Wo,
    bf16_t* __restrict__ wT) {
    int idx = blockIdx.x * 256 + threadIdx.x;          // 4*512*512 threads
    int w   = idx >> 18;
    int rem = idx & ((1 << 18) - 1);
    int kk  = rem >> 9;
    int c   = rem & 511;
    const float* W = (w == 0) ? Wq : (w == 1) ? Wk : (w == 2) ? Wv : Wo;
    wT[(size_t)w*HID*HID + (size_t)c*HID + kk] = (__bf16)W[(size_t)kk*HID + c];
}

// ---------- K1: fused QKV projection (WMMA bf16) ----------
// q,k: [B][H][N][D] bf16 ; vT: [B][H][D][N] bf16
__global__ void __launch_bounds__(128) k_qkv(
    const float* __restrict__ x, const bf16_t* __restrict__ wT,
    const float* __restrict__ bq, const float* __restrict__ bk, const float* __restrict__ bv,
    bf16_t* __restrict__ q, bf16_t* __restrict__ kout, bf16_t* __restrict__ vT) {
    int wave = threadIdx.x >> 5, lane = threadIdx.x & 31;
    int lm = lane & 15, kh = lane >> 4;
    int ctile = blockIdx.x * 4 + wave;      // 0..95
    int mat   = ctile / 32;                 // 0=q 1=k 2=v
    int cb    = (ctile % 32) * 16;          // output column base (0..496)
    int mbase = blockIdx.y * 16;            // row tile base in M = B*N = 4096

    const float*  xrow = x  + (size_t)(mbase + lm) * HID;
    const bf16_t* wcol = wT + (size_t)mat * HID * HID + (size_t)(cb + lm) * HID;

    v8f acc = {0,0,0,0,0,0,0,0};
    for (int kc = 0; kc < HID; kc += 32) {
        v16bf a  = a_frag_f32(xrow + kc, kh);
        v16bf bf = b_frag_bf16(wcol + kc, kh);
        acc = wmma_bf16(a, bf, acc);
    }
    const float* biasv = (mat == 0) ? bq : (mat == 1) ? bk : bv;
    float bias = biasv[cb + lm];
    int cg = cb + lm, h = cg >> 6, d = cg & 63;

    if (mat < 2) {
        bf16_t* dst = (mat == 0) ? q : kout;
        #pragma unroll
        for (int r = 0; r < 8; ++r) {
            int m2 = mbase + r + 8 * kh;           // C layout: M = r + 8*(lane/16)
            int b  = m2 >> 9, n = m2 & 511;
            dst[((size_t)(b*NH + h)*NSEQ + n)*HD + d] = (__bf16)(acc[r] + bias);
        }
    } else {
        int b  = mbase >> 9;
        int n0 = (mbase & 511) + 8 * kh;           // 8 consecutive n -> packed store
        v8bf p;
        #pragma unroll
        for (int r = 0; r < 8; ++r) p[r] = (__bf16)(acc[r] + bias);
        *reinterpret_cast<v8bf*>(vT + ((size_t)(b*NH + h)*HD + d)*NSEQ + n0) = p;
    }
}

// ---------- K2: distance bias + key-mask fold ----------
// biasM[b][h][qy][kx] = mask[b][kx] ? dist[b][qy][kx][:] . Wd[:,h] + bd[h] : -1e9
__global__ void __launch_bounds__(256) k_bias(
    const float* __restrict__ dist, const unsigned char* __restrict__ mask,
    const float* __restrict__ Wd, const float* __restrict__ bd,
    float* __restrict__ biasM) {
    __shared__ float wds[DD * NH];
    __shared__ float bds[NH];
    int tid = threadIdx.x;
    for (int i = tid; i < DD * NH; i += 256) wds[i] = Wd[i];
    if (tid < NH) bds[tid] = bd[tid];
    __syncthreads();

    size_t idx = (size_t)blockIdx.x * 256 + tid;   // b*N*N + qy*N + kx
    int b   = (int)(idx >> 18);                    // N*N = 262144
    int rem = (int)(idx & 262143);
    int qy  = rem >> 9, kx = rem & 511;
    size_t orow = (size_t)b * NH * NSEQ * NSEQ + (size_t)qy * NSEQ + kx;

    if (!mask[b * NSEQ + kx]) {
        #pragma unroll
        for (int h = 0; h < NH; ++h) biasM[orow + (size_t)h * NSEQ * NSEQ] = -1e9f;
        return;
    }
    const float4* row = reinterpret_cast<const float4*>(dist + idx * DD);
    float acc[NH];
    #pragma unroll
    for (int h = 0; h < NH; ++h) acc[h] = bds[h];
    #pragma unroll
    for (int i = 0; i < 16; ++i) {
        float4 v = row[i];
        float vv[4] = {v.x, v.y, v.z, v.w};
        #pragma unroll
        for (int c4 = 0; c4 < 4; ++c4) {
            float xv = vv[c4];
            const float* wr = &wds[(i * 4 + c4) * NH];
            #pragma unroll
            for (int h = 0; h < NH; ++h) acc[h] += xv * wr[h];
        }
    }
    #pragma unroll
    for (int h = 0; h < NH; ++h) biasM[orow + (size_t)h * NSEQ * NSEQ] = acc[h];
}

// ---------- K3: flash attention (WMMA bf16, online softmax) ----------
// ao[b][n][h*64+d] bf16
__global__ void __launch_bounds__(128) k_attn(
    const bf16_t* __restrict__ q, const bf16_t* __restrict__ kmat,
    const bf16_t* __restrict__ vT, const float* __restrict__ biasM,
    bf16_t* __restrict__ ao) {
    __shared__ bf16_t plds[4][16 * 64];   // per-wave 16x64 P tile (row-major)
    int wave = threadIdx.x >> 5, lane = threadIdx.x & 31;
    int lm = lane & 15, kh = lane >> 4;
    int b = blockIdx.z, h = blockIdx.y;
    int bh = b * NH + h;
    int q0 = blockIdx.x * 64 + wave * 16;

    const bf16_t* qp  = q + ((size_t)bh * NSEQ + q0 + lm) * HD;
    v16bf qa0 = a_frag_bf16(qp,      kh);   // d 0..31
    v16bf qa1 = a_frag_bf16(qp + 32, kh);   // d 32..63

    v8f o[4];
    float mrow[8], lrow[8];
    #pragma unroll
    for (int j = 0; j < 4; ++j) o[j] = (v8f){0,0,0,0,0,0,0,0};
    #pragma unroll
    for (int r = 0; r < 8; ++r) { mrow[r] = -3.0e38f; lrow[r] = 0.0f; }

    bf16_t* pl = &plds[wave][0];
    const float* bbase = biasM + ((size_t)bh * NSEQ + q0 + 8 * kh) * NSEQ;

    for (int kt = 0; kt < NSEQ; kt += 64) {
        if (kt + 64 < NSEQ) __builtin_prefetch(bbase + kt + 64 + lane, 0, 1);
        v8f s[4];
        #pragma unroll
        for (int j = 0; j < 4; ++j) {
            const bf16_t* kp = kmat + ((size_t)bh * NSEQ + kt + j*16 + lm) * HD;
            v16bf b0 = b_frag_bf16(kp,      kh);
            v16bf b1 = b_frag_bf16(kp + 32, kh);
            v8f sj = {0,0,0,0,0,0,0,0};
            sj = wmma_bf16(qa0, b0, sj);
            sj = wmma_bf16(qa1, b1, sj);
            const float* bp = bbase + kt + j * 16 + lm;
            #pragma unroll
            for (int r = 0; r < 8; ++r) sj[r] = sj[r] * 0.125f + bp[(size_t)r * NSEQ];
            s[j] = sj;
        }
        // online softmax: row stats live across the 16 lanes of each half-wave
        float tmax[8];
        #pragma unroll
        for (int r = 0; r < 8; ++r)
            tmax[r] = fmaxf(fmaxf(s[0][r], s[1][r]), fmaxf(s[2][r], s[3][r]));
        #pragma unroll
        for (int off = 8; off >= 1; off >>= 1)
            #pragma unroll
            for (int r = 0; r < 8; ++r)
                tmax[r] = fmaxf(tmax[r], __shfl_xor(tmax[r], off, 32));
        float mnew[8], alpha[8];
        #pragma unroll
        for (int r = 0; r < 8; ++r) {
            mnew[r]  = fmaxf(mrow[r], tmax[r]);
            alpha[r] = __expf(mrow[r] - mnew[r]);
            mrow[r]  = mnew[r];
        }
        float tsum[8];
        #pragma unroll
        for (int r = 0; r < 8; ++r) tsum[r] = 0.0f;
        #pragma unroll
        for (int j = 0; j < 4; ++j)
            #pragma unroll
            for (int r = 0; r < 8; ++r) {
                float p = __expf(s[j][r] - mnew[r]);
                s[j][r] = p;
                tsum[r] += p;
            }
        #pragma unroll
        for (int off = 8; off >= 1; off >>= 1)
            #pragma unroll
            for (int r = 0; r < 8; ++r) tsum[r] += __shfl_xor(tsum[r], off, 32);
        #pragma unroll
        for (int r = 0; r < 8; ++r) lrow[r] = lrow[r] * alpha[r] + tsum[r];
        #pragma unroll
        for (int j = 0; j < 4; ++j)
            #pragma unroll
            for (int r = 0; r < 8; ++r) o[j][r] *= alpha[r];

        // C-layout P -> row-major LDS tile, re-read in A-frag layout
        #pragma unroll
        for (int j = 0; j < 4; ++j)
            #pragma unroll
            for (int r = 0; r < 8; ++r)
                pl[(r + 8 * kh) * 64 + j * 16 + lm] = (__bf16)s[j][r];
        const bf16_t* pr = pl + lm * 64;
        v16bf pa0 = a_frag_bf16(pr,      kh);   // kcol 0..31
        v16bf pa1 = a_frag_bf16(pr + 32, kh);   // kcol 32..63

        #pragma unroll
        for (int j = 0; j < 4; ++j) {
            const bf16_t* vp = vT + ((size_t)bh * HD + j * 16 + lm) * NSEQ + kt;
            v16bf vb0 = b_frag_bf16(vp,      kh);
            v16bf vb1 = b_frag_bf16(vp + 32, kh);
            o[j] = wmma_bf16(pa0, vb0, o[j]);
            o[j] = wmma_bf16(pa1, vb1, o[j]);
        }
    }
    float rinv[8];
    #pragma unroll
    for (int r = 0; r < 8; ++r) rinv[r] = 1.0f / lrow[r];
    #pragma unroll
    for (int j = 0; j < 4; ++j)
        #pragma unroll
        for (int r = 0; r < 8; ++r) {
            int n = q0 + r + 8 * kh;
            ao[((size_t)b * NSEQ + n) * HID + h * HD + j * 16 + lm] =
                (__bf16)(o[j][r] * rinv[r]);
        }
}

// ---------- K4: output projection + final query mask ----------
__global__ void __launch_bounds__(128) k_oproj(
    const bf16_t* __restrict__ ao, const bf16_t* __restrict__ woT,
    const float* __restrict__ bo, const unsigned char* __restrict__ mask,
    float* __restrict__ out) {
    int wave = threadIdx.x >> 5, lane = threadIdx.x & 31;
    int lm = lane & 15, kh = lane >> 4;
    int cb    = (blockIdx.x * 4 + wave) * 16;
    int mbase = blockIdx.y * 16;
    const bf16_t* arow = ao  + (size_t)(mbase + lm) * HID;
    const bf16_t* wcol = woT + (size_t)(cb + lm) * HID;

    v8f acc = {0,0,0,0,0,0,0,0};
    for (int kc = 0; kc < HID; kc += 32) {
        v16bf a  = a_frag_bf16(arow + kc, kh);
        v16bf bf = b_frag_bf16(wcol + kc, kh);
        acc = wmma_bf16(a, bf, acc);
    }
    float bb = bo[cb + lm];
    #pragma unroll
    for (int r = 0; r < 8; ++r) {
        int m2 = mbase + r + 8 * kh;
        int bi = m2 >> 9, n = m2 & 511;
        float mv = mask[bi * NSEQ + n] ? 1.0f : 0.0f;
        out[(size_t)m2 * HID + cb + lm] = (acc[r] + bb) * mv;
    }
}

// ---------- launch ----------
extern "C" void kernel_launch(void* const* d_in, const int* in_sizes, int n_in,
                              void* d_out, int out_size, void* d_ws, size_t ws_size,
                              hipStream_t stream) {
    const float* x    = (const float*)d_in[0];
    const float* dist = (const float*)d_in[1];
    const unsigned char* mask = (const unsigned char*)d_in[2];
    const float* Wq = (const float*)d_in[3];
    const float* bq = (const float*)d_in[4];
    const float* Wk = (const float*)d_in[5];
    const float* bk = (const float*)d_in[6];
    const float* Wv = (const float*)d_in[7];
    const float* bv = (const float*)d_in[8];
    const float* Wo = (const float*)d_in[9];
    const float* bo = (const float*)d_in[10];
    const float* Wd = (const float*)d_in[11];
    const float* bd = (const float*)d_in[12];

    char* ws = (char*)d_ws;
    bf16_t* wT    = (bf16_t*)(ws);               //  2 MB : 4x 512x512 bf16 (transposed)
    bf16_t* q     = (bf16_t*)(ws +  2097152);    //  4 MB : [B][H][N][D]
    bf16_t* kbuf  = (bf16_t*)(ws +  6291456);    //  4 MB : [B][H][N][D]
    bf16_t* vT    = (bf16_t*)(ws + 10485760);    //  4 MB : [B][H][D][N]
    bf16_t* aout  = (bf16_t*)(ws + 14680064);    //  4 MB : [B][N][HID]
    float*  biasM = (float*)(ws + 18874368);     // 64 MB : [B][H][N][N] masked bias
    float*  out   = (float*)d_out;

    k_transpose_w<<<4096, 256, 0, stream>>>(Wq, Wk, Wv, Wo, wT);
    k_qkv<<<dim3(24, 256), 128, 0, stream>>>(x, wT, bq, bk, bv, q, kbuf, vT);
    k_bias<<<8192, 256, 0, stream>>>(dist, mask, Wd, bd, biasM);
    k_attn<<<dim3(8, 8, 8), 128, 0, stream>>>(q, kbuf, vT, biasM, aout);
    k_oproj<<<dim3(8, 256), 128, 0, stream>>>(aout, wT + (size_t)3*HID*HID, bo, mask, out);
}
